// AttnBlock_48911087567725
// MI455X (gfx1250) — matrix operand order repaired and checked by reading, not compiled, not run on hardware
//
#include <hip/hip_runtime.h>

// ---------------------------------------------------------------------------
// AttnBlock for MI455X (gfx1250).
// GroupNorm -> QKV proj -> fused QK^T / softmax(h) / attn*V per (b,w) slice ->
// output proj + residual.  GEMMs: v_wmma_f32_16x16x32_bf16 (wave32, f32 acc).
// Tile staging: global_load_async_to_lds_b128 (ASYNCcnt), double-buffered so
// the async engine streams tile k+1 while XDL computes tile k.
// Fragments: 2x ds_load_b128 each (A row-major-in-K, B stored [n][k]).
// Global layouts: xn,q,k,o transposed [b][hw][c]; v natural [b][c][hw].
// ---------------------------------------------------------------------------

#define B_  4
#define C_  256
#define H_  64
#define W_  64
#define HW_ 4096
#define G_  32
#define EPS_ 1e-5f

typedef unsigned short u16;
typedef __attribute__((ext_vector_type(16))) __bf16 v16bf;
typedef __attribute__((ext_vector_type(8)))  __bf16 v8bf;
typedef __attribute__((ext_vector_type(8)))  float  v8f;

// Native RNE f32->bf16 (single v_cvt instruction).
__device__ __forceinline__ u16 f2bf(float f) {
    union { __bf16 b; u16 s; } x; x.b = (__bf16)f; return x.s;
}

// Async copy 16 bytes global -> LDS (CDNA5 GLOBAL_LOAD_ASYNC_TO_LDS_B128).
// Low 32 bits of a generic pointer to __shared__ are the LDS byte offset.
__device__ __forceinline__ void async_copy16(void* lds_dst, const void* gsrc) {
    unsigned loff = (unsigned)(unsigned long long)lds_dst;
    asm volatile("global_load_async_to_lds_b128 %0, %1, off"
                 :: "v"(loff), "v"(gsrc) : "memory");
}
// Wait until at most N async ops outstanding (in-order completion).
template <int N>
__device__ __forceinline__ void async_wait() {
    asm volatile("s_wait_asynccnt %0" :: "i"(N) : "memory");
}

// A fragment, LDS row-major in K:  A(m,k) = p[m*ld + k].
__device__ __forceinline__ v16bf frag_a_row(const u16* p, int ld, int lane) {
    int m = lane & 15, hk = (lane >> 4) * 8;
    const u16* r = p + m * ld + hk;
    v8bf lo = *(const v8bf*)(r);
    v8bf hi = *(const v8bf*)(r + 16);
    return __builtin_shufflevector(lo, hi, 0, 1, 2, 3, 4, 5, 6, 7,
                                   8, 9, 10, 11, 12, 13, 14, 15);
}

// B fragment, LDS stored [n][k]:  B(k,n) = p[n*ld + k].
__device__ __forceinline__ v16bf frag_b_col(const u16* p, int ld, int lane) {
    int n = lane & 15, k0 = (lane >> 4) * 8;
    const u16* r = p + n * ld + k0;
    v8bf lo = *(const v8bf*)(r);
    v8bf hi = *(const v8bf*)(r + 16);
    return __builtin_shufflevector(lo, hi, 0, 1, 2, 3, 4, 5, 6, 7,
                                   8, 9, 10, 11, 12, 13, 14, 15);
}

__device__ __forceinline__ v8f wmma_bf16(v16bf a, v16bf b, v8f c) {
    return __builtin_amdgcn_wmma_f32_16x16x32_bf16(
        false, a, false, b, (short)0, c, false, false);
}

// ---------------------------------------------------------------------------
// Kernel 1: GroupNorm -> bf16, stored TRANSPOSED [b][hw][c].
// ---------------------------------------------------------------------------
__global__ __launch_bounds__(256) void groupnorm_kernel(
    const float* __restrict__ x, const float* __restrict__ gamma,
    const float* __restrict__ beta, u16* __restrict__ xn_t) {
    __shared__ float rs[256], rs2[256];
    int bg = blockIdx.x;
    int b = bg >> 5, g = bg & 31;
    size_t base = ((size_t)b * C_ + (size_t)g * 8) * HW_;
    size_t tbase = (size_t)b * HW_ * C_;
    int t = threadIdx.x;

    float s = 0.f, s2 = 0.f;
#pragma unroll 4
    for (int i = 0; i < 128; ++i) {
        float v = x[base + t + i * 256];
        s += v; s2 += v * v;
    }
    rs[t] = s; rs2[t] = s2;
    __syncthreads();
    for (int off = 128; off >= 1; off >>= 1) {
        if (t < off) { rs[t] += rs[t + off]; rs2[t] += rs2[t + off]; }
        __syncthreads();
    }
    float mu   = rs[0] * (1.0f / 32768.0f);
    float var  = rs2[0] * (1.0f / 32768.0f) - mu * mu;
    float rstd = rsqrtf(var + EPS_);

#pragma unroll 4
    for (int i = 0; i < 128; ++i) {
        int e = t + i * 256;
        int c = g * 8 + (e >> 12);
        int hw = e & 4095;
        float v = (x[base + e] - mu) * rstd * gamma[c] + beta[c];
        xn_t[tbase + (size_t)hw * C_ + c] = f2bf(v);
    }
}

// ---------------------------------------------------------------------------
// Kernel 2: pack q_w,k_w,v_w,o_w -> bf16.
// ---------------------------------------------------------------------------
__global__ __launch_bounds__(256) void pack_w_kernel(
    const float* __restrict__ qw, const float* __restrict__ kw,
    const float* __restrict__ vw, const float* __restrict__ ow,
    u16* __restrict__ wpack) {
    int i = blockIdx.x * 256 + threadIdx.x;
    int m = i >> 16;
    int r = i & 65535;
    const float* src = (m == 0) ? qw : (m == 1) ? kw : (m == 2) ? vw : ow;
    wpack[i] = f2bf(src[r]);
}

// ---------------------------------------------------------------------------
// Kernel 3: QKV projections, double-buffered async pipeline over 8 k-chunks.
// grid = (HW/64, 3*4, B), block 256.  q,k out transposed; v natural.
// ---------------------------------------------------------------------------
__global__ __launch_bounds__(256) void proj_qkv_kernel(
    const u16* __restrict__ xn_t, const u16* __restrict__ wpack,
    const float* __restrict__ qb, const float* __restrict__ kb,
    const float* __restrict__ vb,
    u16* __restrict__ qo, u16* __restrict__ ko, u16* __restrict__ vo) {
    __shared__ alignas(16) u16 Wl[2][64 * 32];   // [co][ci]  A row-major
    __shared__ alignas(16) u16 Xl[2][64 * 32];   // [hw][ci]  B [n][k]

    int hw0 = blockIdx.x * 64;
    int mo  = blockIdx.y >> 2;
    int co0 = (blockIdx.y & 3) * 64;
    int b   = blockIdx.z;

    const u16*   Wg   = wpack + (size_t)mo * C_ * C_;
    const float* bias = (mo == 0) ? qb : (mo == 1) ? kb : vb;
    u16*         out  = (mo == 0) ? qo : (mo == 1) ? ko : vo;

    int t = threadIdx.x, lane = t & 31, wv = t >> 5;
    int row = t >> 2, cc = t & 3;               // 64 rows x 4 16B chunks

    const u16* wsrc = Wg + (size_t)(co0 + row) * C_ + cc * 8;
    const u16* xsrc = xn_t + ((size_t)b * HW_ + hw0 + row) * C_ + cc * 8;
    u16* wdst = (u16*)Wl + row * 32 + cc * 8;   // buffer-relative, + 2048*buf
    u16* xdst = (u16*)Xl + row * 32 + cc * 8;

    v8f acc[2];
#pragma unroll
    for (int ti = 0; ti < 2; ++ti)
#pragma unroll
        for (int v = 0; v < 8; ++v) acc[ti][v] = 0.f;

    int mt  = wv >> 1;                          // both tiles share A
    int nt0 = (wv * 2) & 3;

    // prologue: chunk 0 into buffer 0
    async_copy16(wdst, wsrc);
    async_copy16(xdst, xsrc);

    for (int kk = 0; kk < 8; ++kk) {
        int cur = kk & 1;
        if (kk < 7) {                           // stream chunk kk+1
            async_copy16(wdst + (cur ^ 1) * 2048, wsrc + (kk + 1) * 32);
            async_copy16(xdst + (cur ^ 1) * 2048, xsrc + (kk + 1) * 32);
            async_wait<2>();                    // chunk kk landed (in-order)
        } else {
            async_wait<0>();
        }
        __syncthreads();
        v16bf a   = frag_a_row(Wl[cur] + mt * 16 * 32, 32, lane);
        v16bf b0  = frag_b_col(Xl[cur] + nt0 * 16 * 32, 32, lane);
        v16bf b1  = frag_b_col(Xl[cur] + (nt0 + 1) * 16 * 32, 32, lane);
        acc[0] = wmma_bf16(a, b0, acc[0]);
        acc[1] = wmma_bf16(a, b1, acc[1]);
        __syncthreads();
    }

    int n = lane & 15, mh = (lane >> 4) * 8;
#pragma unroll
    for (int ti = 0; ti < 2; ++ti) {
#pragma unroll
        for (int v = 0; v < 8; ++v) {
            int c  = co0 + mt * 16 + v + mh;
            int hw = hw0 + (nt0 + ti) * 16 + n;
            size_t idx = (mo == 2) ? ((size_t)b * C_ + c) * HW_ + hw
                                   : ((size_t)b * HW_ + hw) * C_ + c;
            out[idx] = f2bf(acc[ti][v] + bias[c]);
        }
    }
}

// ---------------------------------------------------------------------------
// Kernel 4: fused attention, one (b,w) slice per block.
// K and V tiles for a column block are issued together; V streams in behind
// the S-GEMM + softmax.  LDS: 32(Q)+16(K)+16(V)+8(S)+4(attn)+1 = 77 KB.
// ---------------------------------------------------------------------------
__global__ __launch_bounds__(256) void attn_kernel(
    const u16* __restrict__ qT, const u16* __restrict__ kT,
    const u16* __restrict__ v, u16* __restrict__ oT) {
    __shared__ alignas(16) u16   Qs[64 * C_];   // [h][c]      32 KB
    __shared__ alignas(16) u16   Kb[32 * C_];   // [col][c]    16 KB
    __shared__ alignas(16) u16   Vb[C_ * 32];   // [c][col]    16 KB
    __shared__ alignas(16) float Ss[64 * 32];   // [h][col]     8 KB
    __shared__ alignas(16) u16   As[64 * 32];   // [h][col]     4 KB
    __shared__ float red[8 * 32];               //              1 KB

    int w = blockIdx.x, b = blockIdx.y;
    int t = threadIdx.x, lane = t & 31, wv = t >> 5;
    const float scale = 0.0625f;                // 1/sqrt(256)

    // ---- stage Q slice (64 x 256) via async ----
#pragma unroll
    for (int i = 0; i < 8; ++i) {
        int qc = t + i * 256;
        int r = qc >> 5, c8 = qc & 31;
        async_copy16(Qs + r * 256 + c8 * 8,
                     qT + ((size_t)b * HW_ + r * W_ + w) * C_ + c8 * 8);
    }
    async_wait<0>();
    __syncthreads();

    int smt = wv & 3;                           // S tile: h-tile
    int snt = wv >> 2;                          // S tile: col-tile
    int col = t & 31, seg = t >> 5;             // softmax mapping

    // per-thread staging addresses (4 chunks each for K and V)
    int kr = t >> 6;                            // reuse below per-i
    (void)kr;

    // Hoisted Q A-fragments (invariant over all 128 column tiles).
    v16bf qa[8];
#pragma unroll
    for (int kk = 0; kk < 8; ++kk)
        qa[kk] = frag_a_row(Qs + smt * 16 * 256 + kk * 32, 256, lane);

    v8f accO[2][4];
#pragma unroll
    for (int mi = 0; mi < 2; ++mi)
#pragma unroll
        for (int nt = 0; nt < 4; ++nt)
#pragma unroll
            for (int e = 0; e < 8; ++e) accO[mi][nt][e] = 0.f;

    for (int jb = 0; jb < 128; ++jb) {
        int col0 = jb * 32;

        // ---- issue K tile then V tile (V overlaps S-GEMM + softmax) ----
#pragma unroll
        for (int i = 0; i < 4; ++i) {
            int qc = t + i * 256;
            int r = qc >> 5, c8 = qc & 31;
            async_copy16(Kb + r * 256 + c8 * 8,
                         kT + ((size_t)b * HW_ + col0 + r) * C_ + c8 * 8);
        }
#pragma unroll
        for (int i = 0; i < 4; ++i) {
            int qc = t + i * 256;
            int r = qc >> 2, c4 = qc & 3;
            async_copy16(Vb + r * 32 + c4 * 8,
                         v + ((size_t)b * C_ + r) * HW_ + col0 + c4 * 8);
        }
        async_wait<4>();                        // K landed; V still in flight
        if (jb < 127)
            __builtin_prefetch(kT + ((size_t)b * HW_ + col0 + 32) * C_, 0, 3);
        __syncthreads();

        // ---- S = Q^T K : one 16x16 tile per wave, K-dim 256 ----
        v8f s;
#pragma unroll
        for (int e = 0; e < 8; ++e) s[e] = 0.f;
#pragma unroll
        for (int kk = 0; kk < 8; ++kk) {
            v16bf bf = frag_b_col(Kb + snt * 16 * 256 + kk * 32, 256, lane);
            s = wmma_bf16(qa[kk], bf, s);
        }
        {
            int n = lane & 15, mh = (lane >> 4) * 8;
#pragma unroll
            for (int e = 0; e < 8; ++e)
                Ss[(smt * 16 + e + mh) * 32 + snt * 16 + n] = s[e];
        }
        __syncthreads();

        // ---- softmax over h per column (8 segs x 8 rows) ----
        float pmax = -1e30f;
#pragma unroll
        for (int r = 0; r < 8; ++r) pmax = fmaxf(pmax, Ss[(seg * 8 + r) * 32 + col]);
        red[seg * 32 + col] = pmax;
        __syncthreads();
        float gmax = red[col];
#pragma unroll
        for (int s2 = 1; s2 < 8; ++s2) gmax = fmaxf(gmax, red[s2 * 32 + col]);
        __syncthreads();
        float ex[8];
        float psum = 0.f;
#pragma unroll
        for (int r = 0; r < 8; ++r) {
            ex[r] = __expf((Ss[(seg * 8 + r) * 32 + col] - gmax) * scale);
            psum += ex[r];
        }
        red[seg * 32 + col] = psum;
        __syncthreads();
        float tot = 0.f;
#pragma unroll
        for (int s2 = 0; s2 < 8; ++s2) tot += red[s2 * 32 + col];
        float inv = 1.0f / tot;
#pragma unroll
        for (int r = 0; r < 8; ++r)
            As[(seg * 8 + r) * 32 + col] = f2bf(ex[r] * inv);

        async_wait<0>();                        // V tile landed
        __syncthreads();                        // As + Vb visible to all

        // ---- O += V * attn^T : share B fragment across both c-tiles ----
        v16bf a0 = frag_a_row(Vb + (wv * 2 + 0) * 16 * 32, 32, lane);
        v16bf a1 = frag_a_row(Vb + (wv * 2 + 1) * 16 * 32, 32, lane);
#pragma unroll
        for (int nt = 0; nt < 4; ++nt) {
            v16bf bf = frag_b_col(As + nt * 16 * 32, 32, lane);
            accO[0][nt] = wmma_bf16(a0, bf, accO[0][nt]);
            accO[1][nt] = wmma_bf16(a1, bf, accO[1][nt]);
        }
        __syncthreads();                        // protect tiles for next iter
    }

    // ---- epilogue: oT[b, h*W + w, c] ----
    int n = lane & 15, mh = (lane >> 4) * 8;
#pragma unroll
    for (int mi = 0; mi < 2; ++mi) {
        int cbase = (wv * 2 + mi) * 16;
#pragma unroll
        for (int nt = 0; nt < 4; ++nt)
#pragma unroll
            for (int e = 0; e < 8; ++e) {
                int c = cbase + e + mh;
                int h = nt * 16 + n;
                oT[((size_t)b * HW_ + (size_t)h * W_ + w) * C_ + c] =
                    f2bf(accO[mi][nt][e]);
            }
    }
}

// ---------------------------------------------------------------------------
// Kernel 5: out = x + o_w * o + o_b, double-buffered async pipeline.
// ---------------------------------------------------------------------------
__global__ __launch_bounds__(256) void final_proj_kernel(
    const u16* __restrict__ oT, const u16* __restrict__ ow,
    const float* __restrict__ ob, const float* __restrict__ x,
    float* __restrict__ out) {
    __shared__ alignas(16) u16 Wl[2][64 * 32];
    __shared__ alignas(16) u16 Xl[2][64 * 32];

    int hw0 = blockIdx.x * 64;
    int co0 = blockIdx.y * 64;
    int b   = blockIdx.z;
    int t = threadIdx.x, lane = t & 31, wv = t >> 5;
    int row = t >> 2, cc = t & 3;

    const u16* wsrc = ow + (size_t)(co0 + row) * C_ + cc * 8;
    const u16* xsrc = oT + ((size_t)b * HW_ + hw0 + row) * C_ + cc * 8;
    u16* wdst = (u16*)Wl + row * 32 + cc * 8;
    u16* xdst = (u16*)Xl + row * 32 + cc * 8;

    v8f acc[2];
#pragma unroll
    for (int ti = 0; ti < 2; ++ti)
#pragma unroll
        for (int v = 0; v < 8; ++v) acc[ti][v] = 0.f;

    int mt  = wv >> 1;
    int nt0 = (wv * 2) & 3;

    async_copy16(wdst, wsrc);
    async_copy16(xdst, xsrc);

    for (int kk = 0; kk < 8; ++kk) {
        int cur = kk & 1;
        if (kk < 7) {
            async_copy16(wdst + (cur ^ 1) * 2048, wsrc + (kk + 1) * 32);
            async_copy16(xdst + (cur ^ 1) * 2048, xsrc + (kk + 1) * 32);
            async_wait<2>();
        } else {
            async_wait<0>();
        }
        __syncthreads();
        v16bf a  = frag_a_row(Wl[cur] + mt * 16 * 32, 32, lane);
        v16bf b0 = frag_b_col(Xl[cur] + nt0 * 16 * 32, 32, lane);
        v16bf b1 = frag_b_col(Xl[cur] + (nt0 + 1) * 16 * 32, 32, lane);
        acc[0] = wmma_bf16(a, b0, acc[0]);
        acc[1] = wmma_bf16(a, b1, acc[1]);
        __syncthreads();
    }

    int n = lane & 15, mh = (lane >> 4) * 8;
#pragma unroll
    for (int ti = 0; ti < 2; ++ti) {
#pragma unroll
        for (int v = 0; v < 8; ++v) {
            int c  = co0 + mt * 16 + v + mh;
            int hw = hw0 + (nt0 + ti) * 16 + n;
            size_t idx = ((size_t)b * C_ + c) * HW_ + hw;
            out[idx] = x[idx] + acc[ti][v] + ob[c];
        }
    }
}

// ---------------------------------------------------------------------------
// Host launcher.
// Inputs: x, gamma, beta, q_w, q_b, k_w, k_b, v_w, v_b, o_w, o_b
// Workspace (bf16): xn_t | wpack | qT | kT | v | oT  ~= 40.5 MB
// ---------------------------------------------------------------------------
extern "C" void kernel_launch(void* const* d_in, const int* in_sizes, int n_in,
                              void* d_out, int out_size, void* d_ws, size_t ws_size,
                              hipStream_t stream) {
    (void)in_sizes; (void)n_in; (void)out_size; (void)ws_size;

    const float* x     = (const float*)d_in[0];
    const float* gamma = (const float*)d_in[1];
    const float* beta  = (const float*)d_in[2];
    const float* q_w   = (const float*)d_in[3];
    const float* q_b   = (const float*)d_in[4];
    const float* k_w   = (const float*)d_in[5];
    const float* k_b   = (const float*)d_in[6];
    const float* v_w   = (const float*)d_in[7];
    const float* v_b   = (const float*)d_in[8];
    const float* o_w   = (const float*)d_in[9];
    const float* o_b   = (const float*)d_in[10];
    float* out = (float*)d_out;

    const size_t NACT = (size_t)B_ * C_ * HW_;   // 4,194,304
    const size_t NW   = (size_t)4 * C_ * C_;     // 262,144
    u16* ws16   = (u16*)d_ws;
    u16* xn_t   = ws16;
    u16* wpack  = xn_t + NACT;
    u16* qT     = wpack + NW;
    u16* kT     = qT + NACT;
    u16* vbuf   = kT + NACT;
    u16* oT     = vbuf + NACT;

    groupnorm_kernel<<<dim3(B_ * G_), dim3(256), 0, stream>>>(x, gamma, beta, xn_t);

    pack_w_kernel<<<dim3((unsigned)(NW / 256)), dim3(256), 0, stream>>>(
        q_w, k_w, v_w, o_w, wpack);

    proj_qkv_kernel<<<dim3(HW_ / 64, 3 * (C_ / 64), B_), dim3(256), 0, stream>>>(
        xn_t, wpack, q_b, k_b, v_b, qT, kT, vbuf);

    attn_kernel<<<dim3(W_, B_), dim3(256), 0, stream>>>(qT, kT, vbuf, oT);

    final_proj_kernel<<<dim3(HW_ / 64, C_ / 64, B_), dim3(256), 0, stream>>>(
        oT, wpack + (size_t)3 * C_ * C_, o_b, x, out);
}
